// RoutedFFN_51333449122352
// MI455X (gfx1250) — matrix operand order repaired and checked by reading, not compiled, not run on hardware
//
#include <hip/hip_runtime.h>

// Problem constants (fixed by the reference)
#define IN_F   1024
#define OUT_F  4096
#define BLK    256
#define NBLK   16
#define NTOK   16384   // 4 * 4096 tokens
#define TM     16      // tokens per workgroup tile

typedef __attribute__((ext_vector_type(8)))  __bf16 v8bf;
typedef __attribute__((ext_vector_type(16))) __bf16 v16bf;
typedef __attribute__((ext_vector_type(8)))  float  v8f;

// ---------------------------------------------------------------------------
// K0: convert w1 and w2 (fp32) to bf16 in workspace (each 4096*1024 elements)
// ---------------------------------------------------------------------------
__global__ __launch_bounds__(256) void convert_kernel(
    const float* __restrict__ w1, const float* __restrict__ w2,
    unsigned short* __restrict__ w1u, unsigned short* __restrict__ w2u)
{
    size_t i = (size_t)blockIdx.x * 256 + threadIdx.x;   // grid covers 4194304
    __bf16* w1b = reinterpret_cast<__bf16*>(w1u);
    __bf16* w2b = reinterpret_cast<__bf16*>(w2u);
    w1b[i] = (__bf16)w1[i];
    w2b[i] = (__bf16)w2[i];
}

// ---------------------------------------------------------------------------
// K1: router. One wave32 per token. Softmax is monotonic -> top-4 of
// (x @ router_w^T + router_b) directly. Emits a 16-bit block mask per token.
// ---------------------------------------------------------------------------
__global__ __launch_bounds__(256) void router_kernel(
    const float* __restrict__ x, const float* __restrict__ rw,
    const float* __restrict__ rb, unsigned* __restrict__ maskout)
{
    const int lane = threadIdx.x & 31;
    const int wave = threadIdx.x >> 5;
    const int t    = blockIdx.x * 8 + wave;   // 2048 blocks * 8 waves = 16384

    float acc[NBLK];
    #pragma unroll
    for (int b = 0; b < NBLK; ++b) acc[b] = 0.0f;

    // 1024 dims = 8 iterations * 32 lanes * float4
    for (int j = 0; j < 8; ++j) {
        const int d = (j * 32 + lane) * 4;
        float4 xv = *(const float4*)(x + (size_t)t * IN_F + d);
        #pragma unroll
        for (int b = 0; b < NBLK; ++b) {
            float4 wv = *(const float4*)(rw + (size_t)b * IN_F + d);
            acc[b] += xv.x * wv.x + xv.y * wv.y + xv.z * wv.z + xv.w * wv.w;
        }
    }
    // wave32 tree reduction per block
    #pragma unroll
    for (int b = 0; b < NBLK; ++b) {
        float v = acc[b];
        #pragma unroll
        for (int off = 16; off >= 1; off >>= 1)
            v += __shfl_xor(v, off, 32);
        acc[b] = v + rb[b];
    }
    if (lane == 0) {
        unsigned m = 0;
        #pragma unroll
        for (int kk = 0; kk < 4; ++kk) {
            int bi = 0; float best = -3.4e38f;
            #pragma unroll
            for (int b = 0; b < NBLK; ++b) {
                bool taken = (m >> b) & 1u;
                if (!taken && acc[b] > best) { best = acc[b]; bi = b; }
            }
            m |= 1u << bi;
        }
        maskout[t] = m;
    }
}

// ---------------------------------------------------------------------------
// K2: fused masked FFN. 8 waves per 16-token tile.
//   h_b = relu(mask(x @ w1_b^T + b1_b))  (bf16 WMMA, f32 accum) -> LDS bf16
//   y  += h_b @ w2_b^T                   (bf16 WMMA, f32 accum in VGPRs)
// ---------------------------------------------------------------------------
__global__ __launch_bounds__(256) void ffn_kernel(
    const float* __restrict__ x,
    const unsigned short* __restrict__ w1u, const float* __restrict__ b1,
    const unsigned short* __restrict__ w2u, const float* __restrict__ b2,
    const unsigned* __restrict__ maskbuf, float* __restrict__ out)
{
    __shared__ __bf16 ldsx[TM * IN_F];  // 32 KB: x tile in bf16
    __shared__ __bf16 hbuf[TM * BLK];   //  8 KB: h tile for current block

    const __bf16* w1b = reinterpret_cast<const __bf16*>(w1u);
    const __bf16* w2b = reinterpret_cast<const __bf16*>(w2u);

    const int tid  = threadIdx.x;
    const int wave = tid >> 5;
    const int lane = tid & 31;
    const int half = lane >> 4;   // 0: lanes 0-15, 1: lanes 16-31
    const int l16  = lane & 15;
    const int t0   = blockIdx.x * TM;

    // ---- stage 0: x tile -> LDS (fp32 -> bf16), fully coalesced ----
    #pragma unroll
    for (int i = 0; i < (TM * IN_F) / 256; ++i) {
        const int idx = i * 256 + tid;
        ldsx[idx] = (__bf16)x[(size_t)t0 * IN_F + idx];
    }
    __syncthreads();

    // Per-lane row masks: C/D tile row M = r + 8*half
    unsigned rowmask[8];
    #pragma unroll
    for (int r = 0; r < 8; ++r) rowmask[r] = maskbuf[t0 + r + 8 * half];

    // y accumulators: wave owns columns [wave*128, wave*128+128) -> 8 tiles
    v8f yacc[8];
    {
        const v8f z = {0.f, 0.f, 0.f, 0.f, 0.f, 0.f, 0.f, 0.f};
        #pragma unroll
        for (int q = 0; q < 8; ++q) yacc[q] = z;
    }

    const int wloc = wave * 32;     // FC1: this wave's h columns within block
    const int d0b  = wave * 128;    // FC2: this wave's y column base

    #pragma unroll 1
    for (int b = 0; b < NBLK; ++b) {
        // ---------------- FC1: two 16x16 h tiles, K = 1024 ----------------
        v8f acc0 = {0.f, 0.f, 0.f, 0.f, 0.f, 0.f, 0.f, 0.f};
        v8f acc1 = {0.f, 0.f, 0.f, 0.f, 0.f, 0.f, 0.f, 0.f};
        for (int k0 = 0; k0 < IN_F; k0 += 32) {
            // A (16x32 bf16): lane(M=l16) holds 16 contiguous K at 16*half
            const int abase = l16 * IN_F + k0 + half * 16;
            v8bf alo = *(const v8bf*)(ldsx + abase);
            v8bf ahi = *(const v8bf*)(ldsx + abase + 8);
            v16bf a = __builtin_shufflevector(alo, ahi,
                0, 1, 2, 3, 4, 5, 6, 7, 8, 9, 10, 11, 12, 13, 14, 15);
            // B (32x16 bf16): lane(N=l16) holds 16 contiguous K at 16*half
            const __bf16* bp0 =
                w1b + (size_t)(b * BLK + wloc + l16) * IN_F + k0 + half * 16;
            v16bf bv0 = *(const v16bf*)bp0;
            v16bf bv1 = *(const v16bf*)(bp0 + 16 * IN_F);
            acc0 = __builtin_amdgcn_wmma_f32_16x16x32_bf16(
                false, a, false, bv0, (short)0, acc0, false, false);
            acc1 = __builtin_amdgcn_wmma_f32_16x16x32_bf16(
                false, a, false, bv1, (short)0, acc1, false, false);
        }
        // bias + relu + routing mask, write bf16 h tile to LDS
        const float bias0 = b1[b * BLK + wloc + l16];
        const float bias1 = b1[b * BLK + wloc + 16 + l16];
        #pragma unroll
        for (int r = 0; r < 8; ++r) {
            const bool on = (rowmask[r] >> b) & 1u;
            const float h0 = on ? fmaxf(acc0[r] + bias0, 0.f) : 0.f;
            const float h1 = on ? fmaxf(acc1[r] + bias1, 0.f) : 0.f;
            const int M = r + 8 * half;
            hbuf[M * BLK + wloc + l16]      = (__bf16)h0;
            hbuf[M * BLK + wloc + 16 + l16] = (__bf16)h1;
        }
        __syncthreads();

        // ---------------- FC2: yacc += h_b @ w2_b^T, K = 256 --------------
        for (int c0 = 0; c0 < BLK; c0 += 32) {
            const int abase = l16 * BLK + c0 + half * 16;
            v8bf alo = *(const v8bf*)(hbuf + abase);
            v8bf ahi = *(const v8bf*)(hbuf + abase + 8);
            v16bf a = __builtin_shufflevector(alo, ahi,
                0, 1, 2, 3, 4, 5, 6, 7, 8, 9, 10, 11, 12, 13, 14, 15);
            #pragma unroll
            for (int q = 0; q < 8; ++q) {
                const int d = d0b + q * 16 + l16;
                const __bf16* bp =
                    w2b + (size_t)d * OUT_F + b * BLK + c0 + half * 16;
                v16bf bv = *(const v16bf*)bp;
                yacc[q] = __builtin_amdgcn_wmma_f32_16x16x32_bf16(
                    false, a, false, bv, (short)0, yacc[q], false, false);
            }
        }
        __syncthreads();   // before next block overwrites hbuf
    }

    // ---- epilogue: + b2, store f32 (coalesced 64B per 16 lanes) ----
    #pragma unroll
    for (int q = 0; q < 8; ++q) {
        const int d = d0b + q * 16 + l16;
        const float bias = b2[d];
        #pragma unroll
        for (int r = 0; r < 8; ++r) {
            const int M = r + 8 * half;
            out[(size_t)(t0 + M) * IN_F + d] = yacc[q][r] + bias;
        }
    }
}

// ---------------------------------------------------------------------------
// Launcher
// ---------------------------------------------------------------------------
extern "C" void kernel_launch(void* const* d_in, const int* in_sizes, int n_in,
                              void* d_out, int out_size, void* d_ws, size_t ws_size,
                              hipStream_t stream) {
    const float* x        = (const float*)d_in[0];
    const float* router_w = (const float*)d_in[1];
    const float* router_b = (const float*)d_in[2];
    const float* w1       = (const float*)d_in[3];
    const float* b1       = (const float*)d_in[4];
    const float* w2       = (const float*)d_in[5];
    const float* b2       = (const float*)d_in[6];
    float* out            = (float*)d_out;

    char* ws = (char*)d_ws;
    unsigned short* w1u = (unsigned short*)ws;                        // 8 MB
    unsigned short* w2u = (unsigned short*)(ws + ((size_t)8 << 20));  // 8 MB
    unsigned* maskbuf   = (unsigned*)(ws + ((size_t)16 << 20));       // 64 KB

    // K0: weight conversion (4096*1024 elements each)
    convert_kernel<<<(OUT_F * IN_F) / 256, 256, 0, stream>>>(w1, w2, w1u, w2u);
    // K1: router top-4 masks, one wave per token
    router_kernel<<<NTOK / 8, 256, 0, stream>>>(x, router_w, router_b, maskbuf);
    // K2: fused masked FFN, 16 tokens per workgroup
    ffn_kernel<<<NTOK / TM, 256, 0, stream>>>(x, w1u, b1, w2u, b2, maskbuf, out);
}
